// TransformerBlock_8950711845612
// MI455X (gfx1250) — matrix operand order repaired and checked by reading, not compiled, not run on hardware
//
#include <hip/hip_runtime.h>
#include <math.h>

// Problem constants (fixed by the reference)
#define BB      2
#define TSEQ    2048
#define DMODEL  1024
#define NHEADS  16
#define DHEAD   64
#define DFF     4096
#define MROWS   (BB * TSEQ)   // 4096

typedef __attribute__((ext_vector_type(16))) __bf16 v16bf;
typedef __attribute__((ext_vector_type(8)))  float  v8f;
typedef int v4i_vs __attribute__((vector_size(16)));   // matches builtin param pointee

union FragB { v16bf v; unsigned u[8]; };

#if defined(__AMDGCN__) && __has_builtin(__builtin_amdgcn_global_load_async_to_lds_b128)
#define HAVE_ASYNC_LDS 1
#else
#define HAVE_ASYNC_LDS 0
#endif

__device__ __forceinline__ unsigned short f2bf(float f) {
  unsigned u = __float_as_uint(f);
  u += 0x7FFFu + ((u >> 16) & 1u);   // round-to-nearest-even
  return (unsigned short)(u >> 16);
}

// 16-byte global->LDS copy: async (ASYNCcnt) when available, sync fallback.
__device__ __forceinline__ void copy16(const unsigned short* g, unsigned short* l) {
#if HAVE_ASYNC_LDS
  __builtin_amdgcn_global_load_async_to_lds_b128(
      (__attribute__((address_space(1))) v4i_vs*)g,
      (__attribute__((address_space(3))) v4i_vs*)l, 0, 0);
#else
  *(uint4*)l = *(const uint4*)g;
#endif
}

__device__ __forceinline__ void wait_async_copies() {
#if HAVE_ASYNC_LDS
#if __has_builtin(__builtin_amdgcn_s_wait_asynccnt)
  __builtin_amdgcn_s_wait_asynccnt(0);
#else
  asm volatile("s_wait_asynccnt 0x0" ::: "memory");
#endif
#endif
}

// ---------------------------------------------------------------------------
// fp32 -> bf16 elementwise convert
// ---------------------------------------------------------------------------
__global__ __launch_bounds__(256)
void cvt_f32_bf16(const float* __restrict__ in, unsigned short* __restrict__ out, int n) {
  int i = blockIdx.x * 256 + threadIdx.x;
  if (i < n) out[i] = f2bf(in[i]);
}

// ---------------------------------------------------------------------------
// fp32 [K][N] -> bf16 transposed [N][K], 32x32 LDS-tiled (coalesced both ways)
// ---------------------------------------------------------------------------
__global__ __launch_bounds__(256)
void cvt_transpose_bf16(const float* __restrict__ in, unsigned short* __restrict__ out,
                        int K, int N) {
  __shared__ unsigned short t[32][33];
  const int bk = blockIdx.y * 32, bn = blockIdx.x * 32;
  const int tx = threadIdx.x & 31, ty = threadIdx.x >> 5;   // ty: 0..7
#pragma unroll
  for (int e = 0; e < 4; ++e) {
    const int k = ty + e * 8;
    t[tx][k] = f2bf(in[(size_t)(bk + k) * N + bn + tx]);
  }
  __syncthreads();
#pragma unroll
  for (int e = 0; e < 4; ++e) {
    const int n = ty + e * 8;
    out[(size_t)(bn + n) * K + bk + tx] = t[n][tx];
  }
}

// ---------------------------------------------------------------------------
// LayerNorm over last dim (1024), output bf16. One block per row.
// ---------------------------------------------------------------------------
__global__ __launch_bounds__(256)
void ln_to_bf16(const float* __restrict__ x, const float* __restrict__ gam,
                const float* __restrict__ bet, unsigned short* __restrict__ out) {
  const int row = blockIdx.x;
  const int tid = threadIdx.x;
  const float* xr = x + (size_t)row * DMODEL;
  float v[4];
  float s = 0.f;
#pragma unroll
  for (int e = 0; e < 4; ++e) { v[e] = xr[tid + 256 * e]; s += v[e]; }
  __shared__ float red[8];
  __shared__ float stat[2];
#pragma unroll
  for (int off = 16; off > 0; off >>= 1) s += __shfl_xor(s, off);
  if ((tid & 31) == 0) red[tid >> 5] = s;
  __syncthreads();
  if (tid == 0) { float t = 0.f; for (int i = 0; i < 8; ++i) t += red[i]; stat[0] = t * (1.f / DMODEL); }
  __syncthreads();
  const float mean = stat[0];
  float sv = 0.f;
#pragma unroll
  for (int e = 0; e < 4; ++e) { float d = v[e] - mean; sv += d * d; }
#pragma unroll
  for (int off = 16; off > 0; off >>= 1) sv += __shfl_xor(sv, off);
  if ((tid & 31) == 0) red[tid >> 5] = sv;
  __syncthreads();
  if (tid == 0) { float t = 0.f; for (int i = 0; i < 8; ++i) t += red[i]; stat[1] = t * (1.f / DMODEL); }
  __syncthreads();
  const float rstd = rsqrtf(stat[1] + 1e-5f);
  unsigned short* orow = out + (size_t)row * DMODEL;
#pragma unroll
  for (int e = 0; e < 4; ++e) {
    int col = tid + 256 * e;
    orow[col] = f2bf((v[e] - mean) * rstd * gam[col] + bet[col]);
  }
}

// ---------------------------------------------------------------------------
// Tiled bf16 GEMM: C[M,N] = A[M,K] * Bt[N,K]^T + bias [+resid] [gelu]
// A row-major bf16 [M][K]; Bt pre-transposed bf16 [N][K].
// Block: 256 threads (8 waves, 4x2). Block tile 128x128, wave tile 32x64.
// Double-buffered LDS staged with async global->LDS B128 copies.
// ---------------------------------------------------------------------------
template <int OUT_BF16, int GELU, int RESID>
__global__ __launch_bounds__(256)
void gemm_bf16_k(const unsigned short* __restrict__ A,
                 const unsigned short* __restrict__ Bt,
                 const float* __restrict__ bias,
                 const float* __restrict__ resid,
                 void* __restrict__ Cout,
                 int M, int N, int K) {
  __shared__ unsigned short As[2][128][40];   // [buf][m][k] padded
  __shared__ unsigned short Bs[2][128][40];   // [buf][n][k] padded
  const int tid  = threadIdx.x;
  const int lane = tid & 31;
  const int wave = tid >> 5;
  const int g    = lane >> 4;
  const int ln   = lane & 15;
  const int wrow = wave >> 1;       // 0..3 -> 32-row slices
  const int wcol = wave & 1;        // 0..1 -> 64-col slices
  const int blockM = blockIdx.y * 128;
  const int blockN = blockIdx.x * 128;

  v8f z = {};
  v8f acc[2][4];
#pragma unroll
  for (int i = 0; i < 2; ++i)
#pragma unroll
    for (int j = 0; j < 4; ++j) acc[i][j] = z;

  // staging: 512 16B chunks per matrix; 2 chunks each per thread per matrix
  const int c0 = tid, c1 = tid + 256;
  auto stage = [&](int buf, int kk) {
    {
      const int r = c0 >> 2, sgA = (c0 & 3) << 3;
      copy16(A  + (size_t)(blockM + r) * K + kk + sgA, &As[buf][r][sgA]);
      copy16(Bt + (size_t)(blockN + r) * K + kk + sgA, &Bs[buf][r][sgA]);
    }
    {
      const int r = c1 >> 2, sgA = (c1 & 3) << 3;
      copy16(A  + (size_t)(blockM + r) * K + kk + sgA, &As[buf][r][sgA]);
      copy16(Bt + (size_t)(blockN + r) * K + kk + sgA, &Bs[buf][r][sgA]);
    }
  };

  stage(0, 0);
  int buf = 0;
  for (int kk = 0; kk < K; kk += 32) {
    wait_async_copies();
    __syncthreads();                       // staged data visible; prev reads done
    if (kk + 32 < K) stage(buf ^ 1, kk + 32);

    FragB a[2], b[4];
#pragma unroll
    for (int i = 0; i < 2; ++i) {
      const int m = wrow * 32 + i * 16 + ln;
#pragma unroll
      for (int p = 0; p < 8; ++p) {
        const int k0 = (p >> 2) * 16 + g * 8 + (p & 3) * 2;   // A-frag layout
        a[i].u[p] = *(const unsigned*)(&As[buf][m][k0]);
      }
    }
#pragma unroll
    for (int j = 0; j < 4; ++j) {
      const int n = wcol * 64 + j * 16 + ln;
#pragma unroll
      for (int p = 0; p < 8; ++p) {
        const int k0 = (p >> 2) * 16 + g * 8 + (p & 3) * 2;   // same k order as A-frag
        b[j].u[p] = *(const unsigned*)(&Bs[buf][n][k0]);
      }
    }
#pragma unroll
    for (int i = 0; i < 2; ++i)
#pragma unroll
      for (int j = 0; j < 4; ++j)
        acc[i][j] = __builtin_amdgcn_wmma_f32_16x16x32_bf16(
            false, a[i].v, false, b[j].v, (short)0, acc[i][j], false, false);
    buf ^= 1;
  }

  // epilogue
#pragma unroll
  for (int i = 0; i < 2; ++i) {
#pragma unroll
    for (int j = 0; j < 4; ++j) {
      const int col = blockN + wcol * 64 + j * 16 + ln;
      const float bv = bias[col];
#pragma unroll
      for (int r = 0; r < 8; ++r) {
        const int row = blockM + wrow * 32 + i * 16 + r + 8 * g;
        float val = acc[i][j][r] + bv;
        if (GELU) val = 0.5f * val * (1.f + erff(val * 0.70710678118654752f));
        if (RESID) val += resid[(size_t)row * N + col];
        if (OUT_BF16)
          ((unsigned short*)Cout)[(size_t)row * N + col] = f2bf(val);
        else
          ((float*)Cout)[(size_t)row * N + col] = val;
      }
    }
  }
}

// ---------------------------------------------------------------------------
// Flash attention, causal. One wave per (b, h, 32-query tile).
// qkv: [4096, 3072] bf16 rows = [q | k | v]. out: [4096, 1024] bf16.
// ---------------------------------------------------------------------------
__global__ __launch_bounds__(256)
void attn_flash(const unsigned short* __restrict__ qkv,
                unsigned short* __restrict__ out) {
  const int tid  = threadIdx.x;
  const int lane = tid & 31;
  const int wave = tid >> 5;
  const int g    = lane >> 4;
  const int ln   = lane & 15;

  const int tileId = blockIdx.x * 8 + wave;    // 0..2047
  const int qt = tileId & 63;                  // T/32 tiles
  const int bh = tileId >> 6;
  const int h  = bh & 15;
  const int b  = bh >> 4;
  const int qi = qt * 32;

  __shared__ unsigned short Plds[8][32][32];   // per-wave P tile (bf16)

  const unsigned short* qbase = qkv + (size_t)(b * TSEQ) * 3072 + h * DHEAD;
  const unsigned short* kbase = qbase + DMODEL;
  const unsigned short* vbase = qbase + 2 * DMODEL;

  // Q fragments (A layout): 2 row tiles x 2 d-chunks
  FragB qa[2][2];
#pragma unroll
  for (int i = 0; i < 2; ++i)
#pragma unroll
    for (int c = 0; c < 2; ++c)
#pragma unroll
      for (int p = 0; p < 8; ++p) {
        const int d0 = c * 32 + (p >> 2) * 16 + g * 8 + (p & 3) * 2;
        qa[i][c].u[p] = *(const unsigned*)(qbase + (size_t)(qi + i * 16 + ln) * 3072 + d0);
      }

  v8f z = {};
  v8f o[2][4];
  float rmax[2][8], rsum[2][8];
#pragma unroll
  for (int i = 0; i < 2; ++i)
#pragma unroll
    for (int j = 0; j < 4; ++j) o[i][j] = z;
#pragma unroll
  for (int i = 0; i < 2; ++i)
#pragma unroll
    for (int r = 0; r < 8; ++r) { rmax[i][r] = -INFINITY; rsum[i][r] = 0.f; }

  for (int kb = 0; kb <= qi + 31; kb += 32) {
    // prefetch next chunk of K and V rows (speculative; OOB dropped)
    __builtin_prefetch(kbase + (size_t)(kb + 32 + ln) * 3072, 0, 1);
    __builtin_prefetch(vbase + (size_t)(kb + 32 + ln) * 3072, 0, 1);

    // ---- S = Q K^T, 32 keys x 32 query rows ----
    v8f s[2][2] = {{z, z}, {z, z}};
#pragma unroll
    for (int jn = 0; jn < 2; ++jn) {
      const int key = kb + jn * 16 + ln;
#pragma unroll
      for (int c = 0; c < 2; ++c) {
        FragB kf;
#pragma unroll
        for (int p = 0; p < 8; ++p) {
          const int d0 = c * 32 + (p >> 2) * 16 + g * 8 + (p & 3) * 2;
          kf.u[p] = *(const unsigned*)(kbase + (size_t)key * 3072 + d0);
        }
#pragma unroll
        for (int i = 0; i < 2; ++i)     // reuse K fragment for both row tiles
          s[i][jn] = __builtin_amdgcn_wmma_f32_16x16x32_bf16(
              false, qa[i][c].v, false, kf.v, (short)0, s[i][jn], false, false);
      }
    }

    // ---- scale + causal mask + online softmax ----
    const int col0 = kb + ln, col1 = kb + 16 + ln;
#pragma unroll
    for (int i = 0; i < 2; ++i) {
#pragma unroll
      for (int r = 0; r < 8; ++r) {
        const int irow = qi + i * 16 + r + 8 * g;
        float v0 = (col0 <= irow) ? s[i][0][r] * 0.125f : -INFINITY;
        float v1 = (col1 <= irow) ? s[i][1][r] * 0.125f : -INFINITY;
        float tm = fmaxf(v0, v1);
#pragma unroll
        for (int off = 1; off < 16; off <<= 1) tm = fmaxf(tm, __shfl_xor(tm, off));
        const float nm = fmaxf(rmax[i][r], tm);
        const float corr = __expf(rmax[i][r] - nm);
        rmax[i][r] = nm;
        const float p0 = __expf(v0 - nm);
        const float p1 = __expf(v1 - nm);
        float ts = p0 + p1;
#pragma unroll
        for (int off = 1; off < 16; off <<= 1) ts += __shfl_xor(ts, off);
        rsum[i][r] = rsum[i][r] * corr + ts;
#pragma unroll
        for (int j = 0; j < 4; ++j) o[i][j][r] *= corr;
        Plds[wave][i * 16 + r + 8 * g][ln]      = f2bf(p0);
        Plds[wave][i * 16 + r + 8 * g][16 + ln] = f2bf(p1);
      }
    }
    asm volatile("" ::: "memory");  // keep wave-local LDS P stores before reads

    // ---- P fragments (A layout 16x32) ----
    FragB pa[2];
#pragma unroll
    for (int i = 0; i < 2; ++i)
#pragma unroll
      for (int p = 0; p < 8; ++p) {
        const int k0 = (p >> 2) * 16 + g * 8 + (p & 3) * 2;
        pa[i].u[p] = *(const unsigned*)(&Plds[wave][i * 16 + ln][k0]);
      }

    // ---- O += P * V (4 d-tiles of 16; V fragment reused for both row tiles)
#pragma unroll
    for (int j = 0; j < 4; ++j) {
      FragB vf;
      const int d = j * 16 + ln;
#pragma unroll
      for (int p = 0; p < 8; ++p) {
        const int k0 = (p >> 2) * 16 + g * 8 + (p & 3) * 2;
        const unsigned e0 = vbase[(size_t)(kb + k0)     * 3072 + d];
        const unsigned e1 = vbase[(size_t)(kb + k0 + 1) * 3072 + d];
        vf.u[p] = e0 | (e1 << 16);
      }
#pragma unroll
      for (int i = 0; i < 2; ++i)
        o[i][j] = __builtin_amdgcn_wmma_f32_16x16x32_bf16(
            false, pa[i].v, false, vf.v, (short)0, o[i][j], false, false);
    }
  }

  // ---- normalize and write out (back to [B,T,D] layout) ----
#pragma unroll
  for (int i = 0; i < 2; ++i)
#pragma unroll
    for (int r = 0; r < 8; ++r) {
      const float inv = 1.f / rsum[i][r];
      const int trow = qi + i * 16 + r + 8 * g;
      unsigned short* orow = out + (size_t)(b * TSEQ + trow) * DMODEL + h * DHEAD;
#pragma unroll
      for (int j = 0; j < 4; ++j) orow[j * 16 + ln] = f2bf(o[i][j][r] * inv);
    }
}

// ---------------------------------------------------------------------------
extern "C" void kernel_launch(void* const* d_in, const int* in_sizes, int n_in,
                              void* d_out, int out_size, void* d_ws, size_t ws_size,
                              hipStream_t stream) {
  const float* x     = (const float*)d_in[0];
  // d_in[1] = causal mask (handled analytically)
  const float* ln1_g = (const float*)d_in[2];
  const float* ln1_b = (const float*)d_in[3];
  const float* qkv_w = (const float*)d_in[4];
  const float* qkv_b = (const float*)d_in[5];
  const float* out_w = (const float*)d_in[6];
  const float* out_b = (const float*)d_in[7];
  const float* ln2_g = (const float*)d_in[8];
  const float* ln2_b = (const float*)d_in[9];
  const float* w1    = (const float*)d_in[10];
  const float* b1    = (const float*)d_in[11];
  const float* w2    = (const float*)d_in[12];
  const float* b2    = (const float*)d_in[13];

  size_t off = 0;
  auto alloc = [&](size_t bytes) -> void* {
    void* p = (char*)d_ws + off;
    off += (bytes + 255) & ~(size_t)255;
    return p;
  };
  unsigned short* h1      = (unsigned short*)alloc((size_t)MROWS * DMODEL * 2);
  unsigned short* qkvw_t  = (unsigned short*)alloc((size_t)DMODEL * 3 * DMODEL * 2);
  unsigned short* qkv_act = (unsigned short*)alloc((size_t)MROWS * 3 * DMODEL * 2);
  unsigned short* attn_o  = (unsigned short*)alloc((size_t)MROWS * DMODEL * 2);
  unsigned short* outw_t  = (unsigned short*)alloc((size_t)DMODEL * DMODEL * 2);
  float*          x1      = (float*)alloc((size_t)MROWS * DMODEL * 4);
  unsigned short* h2      = (unsigned short*)alloc((size_t)MROWS * DMODEL * 2);
  unsigned short* w1_t    = (unsigned short*)alloc((size_t)DMODEL * DFF * 2);
  unsigned short* ff      = (unsigned short*)alloc((size_t)MROWS * DFF * 2);
  unsigned short* w2_t    = (unsigned short*)alloc((size_t)DFF * DMODEL * 2);

  dim3 blk(256);
  // weights: fp32 [K][N] -> bf16 [N][K]
  cvt_transpose_bf16<<<dim3(3 * DMODEL / 32, DMODEL / 32), blk, 0, stream>>>(
      qkv_w, qkvw_t, DMODEL, 3 * DMODEL);
  cvt_transpose_bf16<<<dim3(DMODEL / 32, DMODEL / 32), blk, 0, stream>>>(
      out_w, outw_t, DMODEL, DMODEL);
  cvt_transpose_bf16<<<dim3(DFF / 32, DMODEL / 32), blk, 0, stream>>>(
      w1, w1_t, DMODEL, DFF);
  cvt_transpose_bf16<<<dim3(DMODEL / 32, DFF / 32), blk, 0, stream>>>(
      w2, w2_t, DFF, DMODEL);

  // h1 = LN1(x) -> bf16
  ln_to_bf16<<<dim3(MROWS), blk, 0, stream>>>(x, ln1_g, ln1_b, h1);

  // qkv_act = h1 @ qkv_w + qkv_b  (bf16 out)     [4096 x 3072]
  gemm_bf16_k<1, 0, 0><<<dim3(3 * DMODEL / 128, MROWS / 128), blk, 0, stream>>>(
      h1, qkvw_t, qkv_b, nullptr, qkv_act, MROWS, 3 * DMODEL, DMODEL);

  // attention -> attn_o (bf16)                   [4096 x 1024]
  attn_flash<<<dim3(MROWS * NHEADS / 32 / 8), blk, 0, stream>>>(qkv_act, attn_o);

  // x1 = x + attn_o @ out_w + out_b  (fp32 out)  [4096 x 1024]
  gemm_bf16_k<0, 0, 1><<<dim3(DMODEL / 128, MROWS / 128), blk, 0, stream>>>(
      attn_o, outw_t, out_b, x, x1, MROWS, DMODEL, DMODEL);

  // h2 = LN2(x1) -> bf16
  ln_to_bf16<<<dim3(MROWS), blk, 0, stream>>>(x1, ln2_g, ln2_b, h2);

  // ff = gelu(h2 @ w1 + b1)  (bf16 out)          [4096 x 4096]
  gemm_bf16_k<1, 1, 0><<<dim3(DFF / 128, MROWS / 128), blk, 0, stream>>>(
      h2, w1_t, b1, nullptr, ff, MROWS, DFF, DMODEL);

  // out = x1 + ff @ w2 + b2  (fp32 out)          [4096 x 1024]
  gemm_bf16_k<0, 0, 1><<<dim3(DMODEL / 128, MROWS / 128), blk, 0, stream>>>(
      ff, w2_t, b2, x1, (float*)d_out, MROWS, DMODEL, DFF);

  (void)in_sizes; (void)n_in; (void)out_size; (void)ws_size;
}
